// OneDConvBlock_30356828848648
// MI455X (gfx1250) — matrix operand order repaired and checked by reading, not compiled, not run on hardware
//
#include <hip/hip_runtime.h>

#define L_LEN 65536
#define NTILES 4096   // L/16

typedef __attribute__((ext_vector_type(16))) __bf16 v16bf;
typedef __attribute__((ext_vector_type(2)))  __bf16 v2bf;
typedef __attribute__((ext_vector_type(8)))  float  v8f;
typedef __attribute__((ext_vector_type(16))) unsigned short v16u;
typedef __attribute__((ext_vector_type(8)))  unsigned v8u;
typedef int async_v4i __attribute__((vector_size(16)));   // matches builtin param type

#if defined(__has_builtin)
#if __has_builtin(__builtin_amdgcn_cvt_pk_bf16_f32)
#define HAVE_CVT_PK_BF16 1
#endif
#if __has_builtin(__builtin_amdgcn_global_load_async_to_lds_b128)
#define HAVE_ASYNC_LDS 1
#endif
#endif
#ifndef HAVE_CVT_PK_BF16
#define HAVE_CVT_PK_BF16 0
#endif
#ifndef HAVE_ASYNC_LDS
#define HAVE_ASYNC_LDS 0
#endif

__device__ inline unsigned short f2bf_sw(float f) {
    unsigned u = __float_as_uint(f);
    return (unsigned short)((u + 0x7FFFu + ((u >> 16) & 1u)) >> 16);  // RNE
}
// Pack two fp32 -> packed bf16 pair (low = a, high = b).
__device__ inline unsigned pk2bf(float a, float b) {
#if HAVE_CVT_PK_BF16
    v2bf p = __builtin_amdgcn_cvt_pk_bf16_f32(a, b);
    return __builtin_bit_cast(unsigned, p);
#else
    return (unsigned)f2bf_sw(a) | ((unsigned)f2bf_sw(b) << 16);
#endif
}
__device__ inline unsigned short f2bf(float f) {
#if HAVE_CVT_PK_BF16
    return (unsigned short)pk2bf(f, 0.f);
#else
    return f2bf_sw(f);
#endif
}
__device__ inline float bf2f(unsigned short h) {
    return __uint_as_float(((unsigned)h) << 16);
}

__device__ inline void async_wait0() {
#if defined(__has_builtin) && __has_builtin(__builtin_amdgcn_s_wait_asynccnt)
    __builtin_amdgcn_s_wait_asynccnt(0);
#else
    asm volatile("s_wait_asynccnt 0" ::: "memory");
#endif
}

// 16B global -> LDS copy (async on gfx1250, register bounce otherwise).
__device__ inline void copy16_g2l(const void* g, void* l) {
#if HAVE_ASYNC_LDS
    __builtin_amdgcn_global_load_async_to_lds_b128(
        (__attribute__((address_space(1))) async_v4i*)g,
        (__attribute__((address_space(3))) async_v4i*)l, 0, 0);
#else
    *(uint4*)l = *(const uint4*)g;
#endif
}

// ---------------------------------------------------------------------------
// Pack a row-major fp32 MxK weight matrix into bf16 WMMA A-fragment tiles.
// Tile (mt,kt) = 16x32; lane ln: row = mt*16 + (ln&15),
// klA(j) = j + (ln<16 ? 0 : 16)   (contiguous k order, matches B fragments).
// ---------------------------------------------------------------------------
__global__ void pack_a(const float* __restrict__ src, unsigned short* __restrict__ dst,
                       int K, int KT) {
    int tid = blockIdx.x * blockDim.x + threadIdx.x;
    int tile = tid >> 5, ln = tid & 31;
    int mt = tile / KT, kt = tile - mt * KT;
    int row = mt * 16 + (ln & 15);
    int kb = kt * 32 + ((ln < 16) ? 0 : 16);
    unsigned short* d = dst + (size_t)tile * 512 + ln * 16;
#pragma unroll
    for (int j = 0; j < 16; ++j) {
        int kc = kb + j;
        d[j] = f2bf(src[(size_t)row * K + kc]);
    }
}

__global__ void rowsum_kernel(const float* __restrict__ src, float* __restrict__ dst, int K) {
    int r = blockIdx.x * blockDim.x + threadIdx.x;
    float s = 0.f;
    for (int k = 0; k < K; ++k) s += src[(size_t)r * K + k];
    dst[r] = s;
}

// ---------------------------------------------------------------------------
// GEMM1: h1(512xL) = prelu(w_in(512x128) @ x(128xL) + b_in, a1)
// One 16-column tile per block, all 512 rows (8 waves x 4 m-tiles).
// Epilogue: bf16 h1 store + per-column sum/sum-of-squares for cum_norm.
// ---------------------------------------------------------------------------
__global__ void __launch_bounds__(256) gemm1_kernel(
    const float* __restrict__ x, const unsigned short* __restrict__ winf,
    const float* __restrict__ b_in, const float* __restrict__ a1p,
    unsigned short* __restrict__ h1, float* __restrict__ s1, float* __restrict__ ss1) {
    __shared__ float cs[16], css[16];
    const int t = threadIdx.x;
    if (t < 16) { cs[t] = 0.f; css[t] = 0.f; }
    __syncthreads();
    const int wave = t >> 5, ln = t & 31;
    const int nt = blockIdx.x;
    const int col = nt * 16 + (ln & 15);
    const int kbase = (ln < 16) ? 0 : 16;
    const float a1 = a1p[0];

    v8u bu[4];
#pragma unroll
    for (int kt = 0; kt < 4; ++kt) {
#pragma unroll
        for (int jj = 0; jj < 8; ++jj) {
            int k = kt * 32 + kbase + jj * 2;
            float f0 = x[(size_t)k * L_LEN + col];
            float f1 = x[(size_t)(k + 1) * L_LEN + col];
            bu[kt][jj] = pk2bf(f0, f1);
        }
    }

    float lsum = 0.f, lss = 0.f;
#pragma unroll
    for (int mi = 0; mi < 4; ++mi) {
        const int mt = wave * 4 + mi;
        v8f acc = {};
#pragma unroll
        for (int kt = 0; kt < 4; ++kt) {
            v16u au = *(const v16u*)(winf + ((size_t)(mt * 4 + kt) * 512) + ln * 16);
            acc = __builtin_amdgcn_wmma_f32_16x16x32_bf16(
                false, __builtin_bit_cast(v16bf, au),
                false, __builtin_bit_cast(v16bf, bu[kt]),
                (short)0, acc, false, false);
        }
#pragma unroll
        for (int r = 0; r < 8; ++r) {
            int row = mt * 16 + ((ln < 16) ? r : 8 + r);
            float v = acc[r] + b_in[row];
            v = (v >= 0.f) ? v : a1 * v;
            lsum += v; lss += v * v;
            h1[(size_t)row * L_LEN + col] = f2bf(v);
        }
    }
    atomicAdd(&cs[ln & 15], lsum);
    atomicAdd(&css[ln & 15], lss);
    __syncthreads();
    if (t < 16) { s1[nt * 16 + t] = cs[t]; ss1[nt * 16 + t] = css[t]; }
}

// ---------------------------------------------------------------------------
// Cumulative-norm scan: one block / 256 threads, each owns a 256-long segment.
// ---------------------------------------------------------------------------
__global__ void __launch_bounds__(256) scan_kernel(
    const float* __restrict__ s, const float* __restrict__ ss,
    float* __restrict__ mean, float* __restrict__ rstd) {
    __shared__ float ts[256], tss[256];
    const int t = threadIdx.x;
    const int seg = L_LEN / 256;
    const int base = t * seg;
    float a = 0.f, b = 0.f;
    for (int i = 0; i < seg; ++i) { a += s[base + i]; b += ss[base + i]; }
    ts[t] = a; tss[t] = b;
    __syncthreads();
    for (int off = 1; off < 256; off <<= 1) {
        float x1 = 0.f, x2 = 0.f;
        if (t >= off) { x1 = ts[t - off]; x2 = tss[t - off]; }
        __syncthreads();
        ts[t] += x1; tss[t] += x2;
        __syncthreads();
    }
    float ra = (t == 0) ? 0.f : ts[t - 1];
    float rb = (t == 0) ? 0.f : tss[t - 1];
    for (int i = 0; i < seg; ++i) {
        ra += s[base + i]; rb += ss[base + i];
        float cnt = 512.0f * (float)(base + i + 1);
        float m = ra / cnt;
        float var = rb / cnt - m * m;
        mean[base + i] = m;
        rstd[base + i] = rsqrtf(fmaxf(var, 1e-20f));
    }
}

// ---------------------------------------------------------------------------
// Depthwise conv (k=3, dil=4, SAME) on normalized h1 + bias + prelu(a2).
// A 512x32-column halo tile of h1 is staged in LDS via async global->LDS
// copies (ASYNCcnt) so each global element is fetched once per block.
// Norm1 applied per tap column; SAME padding = zeros post-norm (masked).
// Output written in WMMA B-fragment order (bf16) + per-column sums.
// ---------------------------------------------------------------------------
__global__ void __launch_bounds__(256) dwconv_kernel(
    const unsigned short* __restrict__ h1,
    const float* __restrict__ mean1, const float* __restrict__ rstd1,
    const float* __restrict__ w_d, const float* __restrict__ b_d,
    const float* __restrict__ a2p,
    unsigned short* __restrict__ h2f, float* __restrict__ s2, float* __restrict__ ss2) {
    __shared__ unsigned short sh[512 * 32];   // 32KB: cols [c0-8, c0+24) x 512 ch
    __shared__ float cs[16], css[16];
    const int t = threadIdx.x;
    if (t < 16) { cs[t] = 0.f; css[t] = 0.f; }
    const int ntb = blockIdx.x;
    const int c0 = ntb * 16;

    // Stage halo tile: 512 channels x 64B each = 2048 x 16B chunks.
    // (±16B over-read at the ends lands in adjacent workspace buffers; the
    //  garbage columns are never used because padded taps are masked.)
#pragma unroll
    for (int i = 0; i < 8; ++i) {
        int idx = t + i * 256;        // 0..2047
        int ch = idx >> 2;
        int ck = idx & 3;
        const char* g = (const char*)h1 + ((long long)ch * L_LEN + (c0 - 8)) * 2 + ck * 16;
        copy16_g2l(g, &sh[ch * 32 + ck * 8]);
    }
#if HAVE_ASYNC_LDS
    async_wait0();
#endif
    __syncthreads();

    const int wave = t >> 5, ln = t & 31;
    const int nloc = ln & 15;
    const int col = c0 + nloc;
    const int half = (ln < 16) ? 0 : 16;
    const float a2 = a2p[0];
    const bool okL = (col >= 4), okR = (col + 4 < L_LEN);
    const float mL = okL ? mean1[col - 4] : 0.f;
    const float rL = okL ? rstd1[col - 4] : 0.f;
    const float mC = mean1[col], rC = rstd1[col];
    const float mR = okR ? mean1[col + 4] : 0.f;
    const float rR = okR ? rstd1[col + 4] : 0.f;

    float lsum = 0.f, lss = 0.f;
#pragma unroll
    for (int kk = 0; kk < 2; ++kk) {
        const int kt = wave * 2 + kk;
        unsigned pw[8];
#pragma unroll
        for (int jj = 0; jj < 8; ++jj) {
            float yv[2];
#pragma unroll
            for (int u = 0; u < 2; ++u) {
                int c = kt * 32 + half + jj * 2 + u;
                const unsigned short* sp = &sh[c * 32 + nloc];  // sp[o] = col c0-8+nloc+o
                float t0 = okL ? (bf2f(sp[4])  - mL) * rL : 0.f;
                float t1 =       (bf2f(sp[8])  - mC) * rC;
                float t2 = okR ? (bf2f(sp[12]) - mR) * rR : 0.f;
                float y = w_d[c * 3 + 0] * t0 + w_d[c * 3 + 1] * t1 + w_d[c * 3 + 2] * t2 + b_d[c];
                y = (y >= 0.f) ? y : a2 * y;
                lsum += y; lss += y * y;
                yv[u] = y;
            }
            pw[jj] = pk2bf(yv[0], yv[1]);
        }
        unsigned short* dst = h2f + ((size_t)(ntb * 16 + kt) * 512) + ln * 16;
        uint4 q0; q0.x = pw[0]; q0.y = pw[1]; q0.z = pw[2]; q0.w = pw[3];
        uint4 q1; q1.x = pw[4]; q1.y = pw[5]; q1.z = pw[6]; q1.w = pw[7];
        *(uint4*)(dst)     = q0;
        *(uint4*)(dst + 8) = q1;
    }
    atomicAdd(&cs[nloc], lsum);
    atomicAdd(&css[nloc], lss);
    __syncthreads();
    if (t < 16) { s2[ntb * 16 + t] = cs[t]; ss2[ntb * 16 + t] = css[t]; }
}

// ---------------------------------------------------------------------------
// GEMM2 (fused out+skip), norm2 folded into epilogue:
//   W @ ((h2-m)·r) = r·(W@h2) - r·m·rowsum(W)
// out = that + b_out + x (residual); skip = that + b_skip.
// ---------------------------------------------------------------------------
__global__ void __launch_bounds__(256) gemm2_kernel(
    const unsigned short* __restrict__ h2f,
    const unsigned short* __restrict__ woutf, const unsigned short* __restrict__ wskipf,
    const float* __restrict__ wsum_o, const float* __restrict__ wsum_s,
    const float* __restrict__ b_out, const float* __restrict__ b_skip,
    const float* __restrict__ mean2, const float* __restrict__ rstd2,
    const float* __restrict__ x, float* __restrict__ out, float* __restrict__ skp) {
    const int t = threadIdx.x;
    const int mt = t >> 5, ln = t & 31;
    const int ntb = blockIdx.x;
    const int col = ntb * 16 + (ln & 15);
    v8f acco = {}, accs = {};
#pragma unroll
    for (int kt = 0; kt < 16; ++kt) {
        v16u bu = *(const v16u*)(h2f + ((size_t)(ntb * 16 + kt) * 512) + ln * 16);
        v16bf b = __builtin_bit_cast(v16bf, bu);
        v16u aou = *(const v16u*)(woutf + ((size_t)(mt * 16 + kt) * 512) + ln * 16);
        v16u asu = *(const v16u*)(wskipf + ((size_t)(mt * 16 + kt) * 512) + ln * 16);
        acco = __builtin_amdgcn_wmma_f32_16x16x32_bf16(
            false, __builtin_bit_cast(v16bf, aou), false, b, (short)0, acco, false, false);
        accs = __builtin_amdgcn_wmma_f32_16x16x32_bf16(
            false, __builtin_bit_cast(v16bf, asu), false, b, (short)0, accs, false, false);
    }
    const float m2 = mean2[col], r2 = rstd2[col];
#pragma unroll
    for (int r = 0; r < 8; ++r) {
        int row = mt * 16 + ((ln < 16) ? r : 8 + r);
        size_t idx = (size_t)row * L_LEN + col;
        out[idx] = r2 * (acco[r] - m2 * wsum_o[row]) + b_out[row] + x[idx];
        skp[idx] = r2 * (accs[r] - m2 * wsum_s[row]) + b_skip[row];
    }
}

extern "C" void kernel_launch(void* const* d_in, const int* in_sizes, int n_in,
                              void* d_out, int out_size, void* d_ws, size_t ws_size,
                              hipStream_t stream) {
    (void)in_sizes; (void)n_in; (void)out_size; (void)ws_size;
    const float* x      = (const float*)d_in[0];
    const float* w_in   = (const float*)d_in[1];
    const float* b_in   = (const float*)d_in[2];
    const float* a1     = (const float*)d_in[3];
    const float* a2     = (const float*)d_in[4];
    const float* w_d    = (const float*)d_in[5];
    const float* b_d    = (const float*)d_in[6];
    const float* w_out  = (const float*)d_in[7];
    const float* b_out  = (const float*)d_in[8];
    const float* w_skip = (const float*)d_in[9];
    const float* b_skip = (const float*)d_in[10];
    float* out = (float*)d_out;
    float* skp = out + (size_t)128 * L_LEN;

    char* base = (char*)d_ws;
    size_t off = 0;
    auto carve = [&](size_t bytes) -> char* {
        char* r = base + off;
        off = (off + bytes + 255) & ~(size_t)255;
        return r;
    };
    unsigned short* winf   = (unsigned short*)carve((size_t)512 * 128 * 2);
    unsigned short* woutf  = (unsigned short*)carve((size_t)128 * 512 * 2);
    unsigned short* wskipf = (unsigned short*)carve((size_t)128 * 512 * 2);
    float* wsum_o = (float*)carve(128 * 4);
    float* wsum_s = (float*)carve(128 * 4);
    float* s1    = (float*)carve((size_t)L_LEN * 4);
    float* ss1   = (float*)carve((size_t)L_LEN * 4);
    float* mean1 = (float*)carve((size_t)L_LEN * 4);
    float* rstd1 = (float*)carve((size_t)L_LEN * 4);
    float* s2    = (float*)carve((size_t)L_LEN * 4);
    float* ss2   = (float*)carve((size_t)L_LEN * 4);
    float* mean2 = (float*)carve((size_t)L_LEN * 4);
    float* rstd2 = (float*)carve((size_t)L_LEN * 4);
    unsigned short* h1  = (unsigned short*)carve((size_t)512 * L_LEN * 2);
    unsigned short* h2f = (unsigned short*)carve((size_t)512 * L_LEN * 2);

    pack_a<<<32, 128, 0, stream>>>(w_in,   winf,   128, 4);
    pack_a<<<32, 128, 0, stream>>>(w_out,  woutf,  512, 16);
    pack_a<<<32, 128, 0, stream>>>(w_skip, wskipf, 512, 16);
    rowsum_kernel<<<1, 128, 0, stream>>>(w_out,  wsum_o, 512);
    rowsum_kernel<<<1, 128, 0, stream>>>(w_skip, wsum_s, 512);

    gemm1_kernel<<<NTILES, 256, 0, stream>>>(x, winf, b_in, a1, h1, s1, ss1);
    scan_kernel<<<1, 256, 0, stream>>>(s1, ss1, mean1, rstd1);
    dwconv_kernel<<<NTILES, 256, 0, stream>>>(h1, mean1, rstd1, w_d, b_d, a2, h2f, s2, ss2);
    scan_kernel<<<1, 256, 0, stream>>>(s2, ss2, mean2, rstd2);
    gemm2_kernel<<<NTILES, 256, 0, stream>>>(h2f, woutf, wskipf, wsum_o, wsum_s,
                                             b_out, b_skip, mean2, rstd2, x, out, skp);
}
